// BondEncoding2D_27788438405801
// MI455X (gfx1250) — compile-verified, matrix operand order checked by reading, not run on hardware
//
#include <hip/hip_runtime.h>
#include <hip/hip_bf16.h>

typedef __attribute__((ext_vector_type(16))) _Float16 v16h;
typedef __attribute__((ext_vector_type(8)))  _Float16 v8h;
typedef __attribute__((ext_vector_type(8)))  float    v8f;
typedef __attribute__((ext_vector_type(4)))  float    v4f;

#define NATOMS   512
#define NPAIRS   (NATOMS * NATOMS)   // 262144
#define DMAX     16
#define NHEADS   32
#define STG_STRIDE 132               // 32 rows x 132 f32 staging (bank-friendly, 16B-aligned rows)

// -------- workspace layout --------
// [0,            32768)  : B packed f16  [16 d][2 ntile][32 lane][16 elem]
// [32768,        34816)  : edge_table f16 [32][32]
#define WS_B_ELEMS (DMAX * 2 * 32 * 16)   // 16384 f16

// Pack W[d] (32x32) into per-lane WMMA-B register order and convert tables to f16.
// 16-bit operand layout (ISA 7.12.2): lane<16 holds K {0..7, 16..23}, lane>=16 holds
// K {8..15, 24..31}; column n = ntile*16 + (lane&15).
__global__ __launch_bounds__(256) void prep_kernel(
    const float* __restrict__ edge_dis_weight,   // [64*32*32] (use first 16*32*32)
    const float* __restrict__ edge_table,        // [32][32] f32
    _Float16* __restrict__ ws_b,                 // [16][2][32][16] f16
    _Float16* __restrict__ ws_et)                // [32][32] f16
{
    int t = blockIdx.x * blockDim.x + threadIdx.x;
    if (t < WS_B_ELEMS) {
        int e    = t & 15;
        int lane = (t >> 4) & 31;
        int nt   = (t >> 9) & 1;
        int d    = t >> 10;
        int kk   = ((lane >> 4) << 3) + (e & 7) + ((e >> 3) << 4);  // K index in [0,32)
        int n    = nt * 16 + (lane & 15);                            // output head
        ws_b[t] = (_Float16)edge_dis_weight[d * 1024 + kk * 32 + n]; // W[d][kk][n]
    } else if (t < WS_B_ELEMS + NHEADS * NHEADS) {
        int i = t - WS_B_ELEMS;
        ws_et[i] = (_Float16)edge_table[i];
    }
}

__global__ __launch_bounds__(256) void bond_main_kernel(
    const int*      __restrict__ spatial_pos,   // [512][512]
    const int*      __restrict__ edge_input,    // [512][512][16]
    const float*    __restrict__ spd_table,     // [64][32] f32
    const _Float16* __restrict__ Bp,            // [16][2][32][16] f16
    const _Float16* __restrict__ etab,          // [32][32] f16
    float*          __restrict__ out_spd,       // [32][262144]
    float*          __restrict__ out_edge)      // [32][262144]
{
    __shared__ float stg[NHEADS * STG_STRIDE];

    const int t   = threadIdx.x;
    const int w   = t >> 5;        // wave id 0..7
    const int L   = t & 31;        // lane
    const int row = L & 15;        // M row within 16-pair tile
    const int hi  = L >> 4;        // half-wave select
    const int wg_base   = blockIdx.x * 128;   // first pair of this workgroup
    const int tile_base = wg_base + w * 16;   // first pair of this wave's tile

    // ---------------- phi_edge: 16-pair x (K=512) x 32 GEMM via WMMA ----------------
    v8f acc0 = {};
    v8f acc1 = {};
    union { v16h v; v8h h[2]; } A, B0, B1;

    const int p_row = tile_base + row;        // this lane's pair (both halves same row)
    #pragma unroll
    for (int d = 0; d < DMAX; ++d) {
        // gather A: edge_table row for this pair/hop, per-lane K slices
        int bond = edge_input[p_row * DMAX + d];
        const _Float16* arow = etab + bond * NHEADS + hi * 8;
        A.h[0] = *(const v8h*)(arow);         // K 0..7  (hi=0) / 8..15  (hi=1)
        A.h[1] = *(const v8h*)(arow + 16);    // K 16..23 (hi=0) / 24..31 (hi=1)
        // B: pre-packed per-lane operand, contiguous 32B
        const _Float16* bbase = Bp + ((d * 2 + 0) * 32 + L) * 16;
        B0.v = *(const v16h*)(bbase);
        B1.v = *(const v16h*)(bbase + 32 * 16);
        acc0 = __builtin_amdgcn_wmma_f32_16x16x32_f16(false, A.v, false, B0.v,
                                                      (short)0, acc0, false, false);
        acc1 = __builtin_amdgcn_wmma_f32_16x16x32_f16(false, A.v, false, B1.v,
                                                      (short)0, acc1, false, false);
    }

    // divide by clamped SPD and stage C tiles into LDS as [head][local_pair]
    #pragma unroll
    for (int r = 0; r < 8; ++r) {
        int pl  = w * 16 + r + hi * 8;                    // local pair in [0,128)
        int sp  = spatial_pos[wg_base + pl];
        float inv = 1.0f / (float)(sp == 0 ? 1 : sp);
        stg[(row)      * STG_STRIDE + pl] = acc0[r] * inv;   // heads 0..15
        stg[(16 + row) * STG_STRIDE + pl] = acc1[r] * inv;   // heads 16..31
    }
    __syncthreads();

    // cooperative coalesced store: each (wave, pass) writes one 128-float head row
    #pragma unroll
    for (int kk = 0; kk < 4; ++kk) {
        int k = kk * 8 + w;
        v4f val = *(const v4f*)&stg[k * STG_STRIDE + L * 4];
        *(v4f*)(out_edge + (size_t)k * NPAIRS + wg_base + L * 4) = val;
    }
    __syncthreads();

    // ---------------- phi_spd: gather + transpose through the same staging ----------------
    {
        int pl   = t & 127;
        int half = t >> 7;
        int sp   = spatial_pos[wg_base + pl];
        const float* srow = spd_table + sp * NHEADS + half * 16;
        #pragma unroll
        for (int h = 0; h < 16; ++h)
            stg[(half * 16 + h) * STG_STRIDE + pl] = srow[h];
    }
    __syncthreads();

    #pragma unroll
    for (int kk = 0; kk < 4; ++kk) {
        int k = kk * 8 + w;
        v4f val = *(const v4f*)&stg[k * STG_STRIDE + L * 4];
        *(v4f*)(out_spd + (size_t)k * NPAIRS + wg_base + L * 4) = val;
    }
}

extern "C" void kernel_launch(void* const* d_in, const int* in_sizes, int n_in,
                              void* d_out, int out_size, void* d_ws, size_t ws_size,
                              hipStream_t stream) {
    const int*   spatial_pos = (const int*)d_in[0];
    const int*   edge_input  = (const int*)d_in[1];
    // d_in[2] = max_dist scalar (compile-time 16 here)
    const float* spd_table   = (const float*)d_in[3];
    const float* edge_table  = (const float*)d_in[4];
    const float* edw         = (const float*)d_in[5];

    _Float16* ws_b  = (_Float16*)d_ws;
    _Float16* ws_et = ws_b + WS_B_ELEMS;

    float* out_spd  = (float*)d_out;                      // phi_spd first in tuple
    float* out_edge = out_spd + (size_t)NHEADS * NPAIRS;  // then phi_edge

    // pack tables (17408 elements)
    prep_kernel<<<(WS_B_ELEMS + NHEADS * NHEADS + 255) / 256, 256, 0, stream>>>(
        edw, edge_table, ws_b, ws_et);

    // 2048 workgroups x 256 threads (8 waves x 16 pairs = 128 pairs per WG)
    bond_main_kernel<<<NPAIRS / 128, 256, 0, stream>>>(
        spatial_pos, edge_input, spd_table, ws_b, ws_et, out_spd, out_edge);
}
